// EIGLayerSimple_67997922230879
// MI455X (gfx1250) — matrix operand order repaired and checked by reading, not compiled, not run on hardware
//
#include <hip/hip_runtime.h>
#include <math.h>

typedef __attribute__((ext_vector_type(16))) __bf16 v16bf;
typedef __attribute__((ext_vector_type(8)))  float  v8f;
typedef __attribute__((ext_vector_type(4)))  unsigned v4u;
typedef __attribute__((ext_vector_type(8)))  int      v8i;
typedef __attribute__((ext_vector_type(4)))  int      v4i;

#define AVG_D_LOG 2.565f
#define EPS_F 1e-5f

static constexpr int N_NODES = 50000;   // 3125 * 16
static constexpr int N_EDGES = 600000;
static constexpr int D = 128;

struct alignas(16) F4 { float x, y, z, w; };
struct alignas(16) U4 { unsigned x, y, z, w; };
struct alignas(8)  B4 { __bf16 e[4]; };

union Frag {
    U4 q[2];
    v16bf v;
    __bf16 e[16];
};

// monotonic float <-> uint mapping for atomic max/min
__device__ __forceinline__ unsigned f2ord(float f) {
    unsigned u = __float_as_uint(f);
    return (u & 0x80000000u) ? ~u : (u | 0x80000000u);
}
__device__ __forceinline__ float ord2f(unsigned u) {
    return (u & 0x80000000u) ? __uint_as_float(u & 0x7fffffffu)
                             : __uint_as_float(~u);
}

// ---------------- fill ----------------
__global__ void k_fill(unsigned* __restrict__ p, unsigned val, long n) {
    long i = (long)blockIdx.x * blockDim.x + threadIdx.x;
    long stride = (long)gridDim.x * blockDim.x;
    for (; i < n; i += stride) p[i] = val;
}

// ---------------- W transpose+convert: W[k][n] f32 -> Wt[n][k] bf16 ----------------
__global__ void k_wt(const float* __restrict__ W, __bf16* __restrict__ Wt) {
    int idx = blockIdx.x * 256 + threadIdx.x;   // 0 .. 1536*128-1
    int k = idx >> 7, n = idx & 127;
    Wt[n * 1536 + k] = (__bf16)W[idx];
}

// ---------------- edge scatter: one wave per edge ----------------
__global__ __launch_bounds__(256) void k_edge(
    const float* __restrict__ h, const int* __restrict__ src,
    const int* __restrict__ dst,
    float* __restrict__ s1, float* __restrict__ s2,
    unsigned* __restrict__ mx, unsigned* __restrict__ mn,
    float* __restrict__ deg)
{
    int e = blockIdx.x * 8 + (threadIdx.x >> 5);
    int lane = threadIdx.x & 31;
    int s = src[e], d = dst[e];
    F4 v = *(const F4*)(h + (long)s * D + lane * 4);
    long base = (long)d * D + lane * 4;
    float vv[4] = {v.x, v.y, v.z, v.w};
#pragma unroll
    for (int i = 0; i < 4; ++i) {
        float x = vv[i];
        unsigned o = f2ord(x);
        atomicAdd(&s1[base + i], x);
        atomicAdd(&s2[base + i], x * x);
        atomicMax(&mx[base + i], o);
        atomicMin(&mn[base + i], o);
    }
    if (lane == 0) atomicAdd(&deg[d], 1.0f);
}

// ---------------- per-node aggregation -> bf16 agg[N,512] + scalers ----------------
__global__ __launch_bounds__(256) void k_node(
    const float* __restrict__ s1, const float* __restrict__ s2,
    const unsigned* __restrict__ mx, const unsigned* __restrict__ mn,
    const float* __restrict__ deg,
    __bf16* __restrict__ agg, float* __restrict__ scal)
{
    int node = blockIdx.x * 8 + (threadIdx.x >> 5);
    int lane = threadIdx.x & 31;
    long base = (long)node * D + lane * 4;

    F4 a1 = *(const F4*)(s1 + base);
    F4 a2 = *(const F4*)(s2 + base);
    U4 xo = *(const U4*)(mx + base);
    U4 no = *(const U4*)(mn + base);

    float dg = deg[node];
    float ds = fmaxf(dg, 1.0f);
    float inv = 1.0f / ds;
    bool has = dg > 0.0f;

    float a1a[4] = {a1.x, a1.y, a1.z, a1.w};
    float a2a[4] = {a2.x, a2.y, a2.z, a2.w};
    unsigned xoa[4] = {xo.x, xo.y, xo.z, xo.w};
    unsigned noa[4] = {no.x, no.y, no.z, no.w};

    float meanv[4], stdv[4], mxv[4], mnv[4];
#pragma unroll
    for (int i = 0; i < 4; ++i) {
        float m = a1a[i] * inv;
        meanv[i] = m;
        float var = a2a[i] * inv - m * m;
        stdv[i] = sqrtf(fmaxf(var, 0.0f) + EPS_F);
        mxv[i] = has ? ord2f(xoa[i]) : 0.0f;
        mnv[i] = has ? ord2f(noa[i]) : 0.0f;
    }

    __bf16* row = agg + (long)node * 512 + lane * 4;
    const float* secs[4] = {meanv, mxv, mnv, stdv};
#pragma unroll
    for (int sidx = 0; sidx < 4; ++sidx) {
        B4 t;
#pragma unroll
        for (int i = 0; i < 4; ++i) t.e[i] = (__bf16)secs[sidx][i];
        *(B4*)(row + sidx * 128) = t;
    }
    if (lane == 0) {
        float logd = logf(ds + 1.0f);
        scal[node * 2 + 0] = logd / AVG_D_LOG;      // amplification
        scal[node * 2 + 1] = AVG_D_LOG / logd;      // attenuation
    }
}

// ---------------- WMMA GEMM: out = hcat @ W + b, *snorm; col stats ----------------
// block = 256 thr = 8 waves; block handles 16 rows x 128 cols; wave = one 16-col tile
// A strip (16 x 512 bf16, contiguous 16 KB in global) is DMA'd into LDS by the
// Tensor Data Mover with pad_interval=256dw/pad_amount=4dw, reproducing the
// bank-conflict-free row stride of 520 bf16.
__global__ __launch_bounds__(256) void k_gemm(
    const __bf16* __restrict__ agg, const float* __restrict__ scal,
    const __bf16* __restrict__ Wt,  const float* __restrict__ bpost,
    const float* __restrict__ snorm,
    float* __restrict__ preout, float* __restrict__ colsum,
    float* __restrict__ colsq)
{
    __shared__ __bf16 As[16 * 520];   // 512 + 8 pad per row -> conflict-free b128

    int tb = blockIdx.x * 16;

    if (threadIdx.x < 32) {
        // ---- TDM descriptor (D#) ----
        unsigned long long ga = (unsigned long long)(const void*)(agg + (long)tb * 512);
        unsigned ldsa = (unsigned)(unsigned long long)(void*)&As[0]; // LDS byte offset
        // group0: count=1 | lds_addr | global_addr(57b) | type=2
        v4u g0 = { 1u, ldsa, (unsigned)ga,
                   (unsigned)((ga >> 32) & 0x01FFFFFFull) | 0x80000000u };
        // group1: data_size=4B(2)<<16 | pad_enable<<20 | pad_interval=7(256dw)<<22
        //         | pad_amount=3(4dw)<<25 ; tensor_dim0=4096 ; tensor_dim1=1 ;
        //         tile_dim0=4096 ; tile_dim1=1 ; stride0=4096 ; stride1=4096
        v8i g1 = { 0x07D20000, 0x10000000, 0x00010000, 0x10000000,
                   0x00000001, 0x00001000, 0x10000000, 0x00000000 };
        v4i gz = { 0, 0, 0, 0 };
#if __clang_major__ >= 23
        v8i gz8 = { 0, 0, 0, 0, 0, 0, 0, 0 };
        __builtin_amdgcn_tensor_load_to_lds(g0, g1, gz, gz, gz8, 0);
#else
        __builtin_amdgcn_tensor_load_to_lds(g0, g1, gz, gz, 0);
#endif
        __builtin_amdgcn_s_wait_tensorcnt(0);
    }
    __syncthreads();

    int wave = threadIdx.x >> 5;
    int lane = threadIdx.x & 31;
    int m  = lane & 15;       // A-row / B-col within tile
    int kh = lane >> 4;       // K-half selector

    float sc1 = scal[(tb + m) * 2 + 0];
    float sc2 = scal[(tb + m) * 2 + 1];

    const __bf16* arow = As + m * 520 + kh * 8;
    const __bf16* wcol = Wt + ((long)(wave * 16 + m)) * 1536 + kh * 8;

    v8f acc = {0.f, 0.f, 0.f, 0.f, 0.f, 0.f, 0.f, 0.f};

    for (int seg = 0; seg < 3; ++seg) {
        float s = (seg == 0) ? 1.0f : ((seg == 1) ? sc1 : sc2);
        const __bf16* wseg = wcol + seg * 512;
        for (int kk = 0; kk < 16; ++kk) {
            Frag a, b;
            const __bf16* ap = arow + kk * 32;
            a.q[0] = *(const U4*)(ap);        // K = kk*32 + kh*8 .. +7
            a.q[1] = *(const U4*)(ap + 16);   // K = +16
            if (seg) {
#pragma unroll
                for (int i = 0; i < 16; ++i)
                    a.e[i] = (__bf16)((float)a.e[i] * s);
            }
            const __bf16* bp = wseg + kk * 32;
            b.q[0] = *(const U4*)(bp);        // K = kk*32 + kh*8 .. +7
            b.q[1] = *(const U4*)(bp + 16);   // K = +16
            acc = __builtin_amdgcn_wmma_f32_16x16x32_bf16(
                false, a.v, false, b.v, (short)0, acc, false, false);
        }
    }

    // epilogue: +bias, *snorm, store pre-BN, column stats
    int colg = wave * 16 + m;
    float bb = bpost[colg];
    float ssum = 0.f, ssq = 0.f;
#pragma unroll
    for (int r = 0; r < 8; ++r) {
        int row = tb + r + ((lane >= 16) ? 8 : 0);
        float v = (acc[r] + bb) * snorm[row];
        preout[(long)row * D + colg] = v;
        ssum += v;
        ssq += v * v;
    }
    atomicAdd(&colsum[colg], ssum);
    atomicAdd(&colsq[colg], ssq);
}

// ---------------- BN stats -> affine per column ----------------
__global__ void k_stats(const float* __restrict__ colsum,
                        const float* __restrict__ colsq,
                        const float* __restrict__ gamma,
                        const float* __restrict__ beta,
                        float* __restrict__ scaleA, float* __restrict__ shiftA)
{
    int c = threadIdx.x;  // 128 threads
    float invN = 1.0f / (float)N_NODES;
    float mu = colsum[c] * invN;
    float var = colsq[c] * invN - mu * mu;
    float sc = gamma[c] * rsqrtf(var + EPS_F);
    scaleA[c] = sc;
    shiftA[c] = beta[c] - mu * sc;
}

// ---------------- BN apply + relu + residual ----------------
__global__ __launch_bounds__(256) void k_final(
    const float* __restrict__ preout, const float* __restrict__ h,
    const float* __restrict__ scaleA, const float* __restrict__ shiftA,
    float* __restrict__ out)
{
    long i = (long)blockIdx.x * 256 + threadIdx.x;
    int c = (int)(i & 127);
    float v = preout[i] * scaleA[c] + shiftA[c];
    v = fmaxf(v, 0.0f);
    out[i] = h[i] + v;
}

extern "C" void kernel_launch(void* const* d_in, const int* in_sizes, int n_in,
                              void* d_out, int out_size, void* d_ws, size_t ws_size,
                              hipStream_t stream) {
    const float* h     = (const float*)d_in[0];
    // d_in[1] = e (unused)
    const float* snorm = (const float*)d_in[2];
    const int*   src   = (const int*)d_in[3];
    const int*   dst   = (const int*)d_in[4];
    const float* W     = (const float*)d_in[5];
    const float* bpost = (const float*)d_in[6];
    const float* gamma = (const float*)d_in[7];
    const float* beta  = (const float*)d_in[8];
    float* out = (float*)d_out;

    char* ws = (char*)d_ws;
    size_t off = 0;
    auto bump = [&](size_t bytes) -> char* {
        char* p = ws + off;
        off = (off + bytes + 255) & ~(size_t)255;
        return p;
    };

    const long ND = (long)N_NODES * D;
    // contiguous zero-span: s1, s2, mx(ordered-min=0), deg, colsum, colsq
    float*    s1     = (float*)bump(ND * 4);
    float*    s2     = (float*)bump(ND * 4);
    unsigned* mx     = (unsigned*)bump(ND * 4);
    float*    deg    = (float*)bump((size_t)N_NODES * 4);
    float*    colsum = (float*)bump(128 * 4);
    float*    colsq  = (float*)bump(128 * 4);
    char*     zero_end = ws + off;
    unsigned* mn     = (unsigned*)bump(ND * 4);
    float*    scal   = (float*)bump((size_t)N_NODES * 2 * 4);
    __bf16*   Wt     = (__bf16*)bump((size_t)1536 * 128 * 2);
    __bf16*   agg    = (__bf16*)bump((size_t)N_NODES * 512 * 2);
    float*    preout = (float*)bump(ND * 4);
    float*    scaleA = (float*)bump(128 * 4);
    float*    shiftA = (float*)bump(128 * 4);

    long zero_cnt = (long)(zero_end - (char*)s1) / 4;
    k_fill<<<2048, 256, 0, stream>>>((unsigned*)s1, 0u, zero_cnt);
    k_fill<<<2048, 256, 0, stream>>>(mn, 0xFFFFFFFFu, ND);   // ordered-max init
    k_wt<<<(1536 * 128) / 256, 256, 0, stream>>>(W, Wt);

    k_edge<<<N_EDGES / 8, 256, 0, stream>>>(h, src, dst, s1, s2, mx, mn, deg);
    k_node<<<N_NODES / 8, 256, 0, stream>>>(s1, s2, mx, mn, deg, agg, scal);
    k_gemm<<<N_NODES / 16, 256, 0, stream>>>(agg, scal, Wt, bpost, snorm,
                                             preout, colsum, colsq);
    k_stats<<<1, 128, 0, stream>>>(colsum, colsq, gamma, beta, scaleA, shiftA);
    k_final<<<(int)(ND / 256), 256, 0, stream>>>(preout, h, scaleA, shiftA, out);
}
